// GraphNeuralEncoder_24335284699305
// MI455X (gfx1250) — matrix operand order repaired and checked by reading, not compiled, not run on hardware
//
#include <hip/hip_runtime.h>
#include <hip/hip_bf16.h>

// Problem constants (match reference)
#define B_   256
#define NC_  99
#define C_   128
#define H_   512
#define L_   3
#define N_   (NC_ + 1)        // 100 nodes/graph
#define NT_  (B_ * N_)        // 25600 rows
#define EPS_ 1e-5f

typedef _Float16 half_t;
typedef __attribute__((ext_vector_type(16))) _Float16 v16h;
typedef __attribute__((ext_vector_type(8)))  float    v8f;

// ---------------------------------------------------------------------------
// f32 -> f16 convert (weights)
// ---------------------------------------------------------------------------
__global__ void cvt_f32_f16_kernel(const float* __restrict__ src,
                                   half_t* __restrict__ dst, int n) {
    int i = blockIdx.x * blockDim.x + threadIdx.x;
    if (i < n) dst[i] = (half_t)src[i];
}

// ---------------------------------------------------------------------------
// Initial node embeddings: x[NT,C] (f32 + f16 copy)
// ---------------------------------------------------------------------------
__global__ void init_embed_kernel(const float* __restrict__ depot_xy,
                                  const float* __restrict__ cust_xy,
                                  const float* __restrict__ demand,
                                  const float* __restrict__ Wd,
                                  const float* __restrict__ bd,
                                  const float* __restrict__ Wi,
                                  const float* __restrict__ bi,
                                  float* __restrict__ x,
                                  half_t* __restrict__ xh) {
    int i = blockIdx.x * blockDim.x + threadIdx.x;   // NT_*C_ exact
    int c    = i & (C_ - 1);
    int node = i >> 7;                                // /C_
    int b    = node / N_;
    int j    = node - b * N_;
    float v;
    if (j == 0) {
        v = depot_xy[b * 2 + 0] * Wd[0 * C_ + c]
          + depot_xy[b * 2 + 1] * Wd[1 * C_ + c] + bd[c];
    } else {
        int q = b * NC_ + (j - 1);
        v = cust_xy[q * 2 + 0] * Wi[0 * C_ + c]
          + cust_xy[q * 2 + 1] * Wi[1 * C_ + c]
          + demand[q]          * Wi[2 * C_ + c] + bi[c];
    }
    x[i]  = v;
    xh[i] = (half_t)v;
}

// ---------------------------------------------------------------------------
// WMMA GEMM: out = A[M,K](f16) x B[K,N](f16), f32 accumulate.
// Block = 256 threads (8 waves), tile 128(M) x 64(N), K-step 32.
// Wave grid 4x2; each wave: 2x2 tiles of v_wmma_f32_16x16x32_f16.
// A tile staged with CDNA5 GLOBAL_LOAD_ASYNC_TO_LDS_B128 (ASYNCcnt path);
// B tile staged transposed via VGPR scatter (async copies cannot transpose).
// EPI: 0 -> store f32 raw (GCN h)
//      1 -> relu(acc + bias) -> f16   (FFN layer 1)
//      2 -> resid + acc + bias -> f32 (FFN layer 2, residual)
// ---------------------------------------------------------------------------
template <int KDIM, int NDIM, int EPI>
__global__ __launch_bounds__(256) void gemm_wmma_kernel(
    const half_t* __restrict__ A, const half_t* __restrict__ Bm,
    const float* __restrict__ bias, const float* __restrict__ resid,
    float* __restrict__ outF, half_t* __restrict__ outH) {
    __shared__ half_t sA[128 * 32];   // [m][k]
    __shared__ half_t sBt[64 * 32];   // [n][k] (transposed stage)

    const int tid  = threadIdx.x;
    const int lane = tid & 31;
    const int wave = tid >> 5;
    const int wm   = wave >> 1;           // 0..3 (M)
    const int wn   = wave & 1;            // 0..1 (N)
    const int l16  = lane & 15;
    const bool hiHalf = (lane & 16) != 0;
    const int blockM = blockIdx.x * 128;
    const int blockN = blockIdx.y * 64;

    v8f acc[2][2] = {};

    const int arow = tid >> 1;            // 0..127
    const int acol = (tid & 1) * 16;      // 0 / 16 halves
    const int brow = tid >> 3;            // 0..31
    const int bcol = (tid & 7) * 8;       // 0..56 halves

    // Per-thread fixed addresses for the A-tile async copy (32B/thread)
    const uint32_t ldsA =
        (uint32_t)(uintptr_t)(&sA[arow * 32 + acol]);   // low 32 bits = LDS offset

    for (int k0 = 0; k0 < KDIM; k0 += 32) {
        // Stage A tile 128x32: direct global -> LDS async copy (no VGPR bounce)
        {
            const uint64_t ga =
                (uint64_t)(uintptr_t)(A + (size_t)(blockM + arow) * KDIM + k0 + acol);
            asm volatile(
                "global_load_async_to_lds_b128 %0, %1, off\n\t"
                "global_load_async_to_lds_b128 %0, %1, off offset:16"
                :: "v"(ldsA), "v"(ga) : "memory");
        }
        // Stage B tile 32x64, transposed into sBt[n][k]
        const uint4* bg =
            (const uint4*)(Bm + (size_t)(k0 + brow) * NDIM + blockN + bcol);
        uint4 bq = bg[0];
        const half_t* bh = (const half_t*)&bq;
#pragma unroll
        for (int i = 0; i < 8; ++i) sBt[(bcol + i) * 32 + brow] = bh[i];

        if (k0 + 32 < KDIM) {  // pull next K-slab of B toward the caches
            __builtin_prefetch(
                (const void*)(Bm + (size_t)(k0 + 32 + brow) * NDIM + blockN + bcol), 0, 1);
        }
        // Drain this wave's async LDS writes, then block barrier
        asm volatile("s_wait_asynccnt 0x0" ::: "memory");
        __syncthreads();

        // Build fragments in contiguous K order (k = 16h + e), same for A and B.
        union { v16h v; half_t h[16]; } af0, af1, bf0, bf1;
        {
            const int koffA = hiHalf ? 16 : 0;  // lanes 16-31 hold K=16..31
            const half_t* pa0 = &sA[(wm * 32 + 0  + l16) * 32];
            const half_t* pa1 = &sA[(wm * 32 + 16 + l16) * 32];
#pragma unroll
            for (int e = 0; e < 8; ++e) {       // elements 0..7 -> K = e (+koffA)
                af0.h[e] = pa0[e + koffA];
                af1.h[e] = pa1[e + koffA];
            }
#pragma unroll
            for (int e = 8; e < 16; ++e) {      // elements 8..15 -> K = e (+koffA)
                af0.h[e] = pa0[e + koffA];
                af1.h[e] = pa1[e + koffA];
            }
            const int koffB = hiHalf ? 16 : 0;  // lanes 16-31 hold K=16..31
            const half_t* pb0 = &sBt[(wn * 32 + 0  + l16) * 32 + koffB];
            const half_t* pb1 = &sBt[(wn * 32 + 16 + l16) * 32 + koffB];
#pragma unroll
            for (int e = 0; e < 16; ++e) {
                bf0.h[e] = pb0[e];
                bf1.h[e] = pb1[e];
            }
        }
        acc[0][0] = __builtin_amdgcn_wmma_f32_16x16x32_f16(
            false, af0.v, false, bf0.v, (short)0, acc[0][0], false, false);
        acc[0][1] = __builtin_amdgcn_wmma_f32_16x16x32_f16(
            false, af0.v, false, bf1.v, (short)0, acc[0][1], false, false);
        acc[1][0] = __builtin_amdgcn_wmma_f32_16x16x32_f16(
            false, af1.v, false, bf0.v, (short)0, acc[1][0], false, false);
        acc[1][1] = __builtin_amdgcn_wmma_f32_16x16x32_f16(
            false, af1.v, false, bf1.v, (short)0, acc[1][1], false, false);
        __syncthreads();
    }

    // Epilogue: C/D layout -> element g: M = g + 8*(lane>=16), N = lane&15
    const int rBase = blockM + wm * 32 + (hiHalf ? 8 : 0);
    const int cBase = blockN + wn * 32 + l16;
#pragma unroll
    for (int tm = 0; tm < 2; ++tm) {
#pragma unroll
        for (int tn = 0; tn < 2; ++tn) {
            const int col = cBase + tn * 16;
#pragma unroll
            for (int g = 0; g < 8; ++g) {
                const int row = rBase + tm * 16 + g;
                const size_t idx = (size_t)row * NDIM + col;
                float v = acc[tm][tn][g];
                if constexpr (EPI == 0) {
                    outF[idx] = v;
                } else if constexpr (EPI == 1) {
                    v += bias[col];
                    outH[idx] = (half_t)(v > 0.f ? v : 0.f);
                } else {
                    outF[idx] = v + bias[col] + resid[idx];
                }
            }
        }
    }
}

// ---------------------------------------------------------------------------
// GCN aggregation via closed-form prefix sum (replaces 1.29M-edge scatter):
//   agg[j] = dis[j] * sum_{i<=j} dis[i]*h[i],  dis[j]=rsqrt(j+1)
//   y[j]   = x[j] + agg[j] + b        (in-place into h is safe: RAW per elem)
// One block per graph, one thread per channel.
// ---------------------------------------------------------------------------
__global__ void gcn_prefix_kernel(const float* __restrict__ h,
                                  const float* __restrict__ x,
                                  const float* __restrict__ bias,
                                  float* __restrict__ y) {
    const int b = blockIdx.x;
    const int c = threadIdx.x;                   // C_ threads
    const size_t base = (size_t)b * N_ * C_ + c;
    const float bc = bias[c];
    float run = 0.f;
    for (int j = 0; j < N_; ++j) {
        const float dj = rsqrtf((float)(j + 1));
        run += dj * h[base + (size_t)j * C_];
        y[base + (size_t)j * C_] = x[base + (size_t)j * C_] + dj * run + bc;
    }
}

// ---------------------------------------------------------------------------
// BatchNorm (training mode) stats per channel -> affine (s, t):
//   norm(v) = v*s[c] + t[c], s = gamma*rsqrt(var+eps), t = beta - mean*s
// One block per channel; no atomics, no workspace zeroing needed.
// ---------------------------------------------------------------------------
__global__ void bn_stats_kernel(const float* __restrict__ y,
                                const float* __restrict__ gamma,
                                const float* __restrict__ beta,
                                float* __restrict__ stats) {
    const int c = blockIdx.x;
    const int t = threadIdx.x;                   // 256 threads
    float s = 0.f, q = 0.f;
    for (int i = t; i < NT_; i += 256) {
        const float v = y[(size_t)i * C_ + c];
        s += v;
        q += v * v;
    }
    __shared__ float ss[256];
    __shared__ float sq[256];
    ss[t] = s; sq[t] = q;
    __syncthreads();
    for (int o = 128; o > 0; o >>= 1) {
        if (t < o) { ss[t] += ss[t + o]; sq[t] += sq[t + o]; }
        __syncthreads();
    }
    if (t == 0) {
        const float mean = ss[0] * (1.0f / NT_);
        const float var  = sq[0] * (1.0f / NT_) - mean * mean;
        const float sc   = gamma[c] * rsqrtf(var + EPS_);
        stats[c]       = sc;
        stats[C_ + c]  = beta[c] - mean * sc;
    }
}

__global__ void bn_apply_kernel(const float* __restrict__ y,
                                const float* __restrict__ stats,
                                float* __restrict__ x,
                                half_t* __restrict__ xh) {
    const int i = blockIdx.x * blockDim.x + threadIdx.x;  // NT_*C_ exact
    const int c = i & (C_ - 1);
    const float v = y[i] * stats[c] + stats[C_ + c];
    x[i]  = v;
    xh[i] = (half_t)v;
}

// ---------------------------------------------------------------------------
// Final: nodes -> d_out[0:NT*C], per-graph mean -> d_out[NT*C : NT*C + B*C]
// ---------------------------------------------------------------------------
__global__ void final_out_kernel(const float* __restrict__ x,
                                 float* __restrict__ out_nodes,
                                 float* __restrict__ out_mean) {
    const int b = blockIdx.x;
    const int c = threadIdx.x;                   // C_ threads
    const size_t base = (size_t)b * N_ * C_ + c;
    float s = 0.f;
    for (int j = 0; j < N_; ++j) {
        const float v = x[base + (size_t)j * C_];
        out_nodes[base + (size_t)j * C_] = v;
        s += v;
    }
    out_mean[(size_t)b * C_ + c] = s * (1.0f / N_);
}

// ---------------------------------------------------------------------------
extern "C" void kernel_launch(void* const* d_in, const int* in_sizes, int n_in,
                              void* d_out, int out_size, void* d_ws, size_t ws_size,
                              hipStream_t stream) {
    (void)in_sizes; (void)n_in; (void)out_size; (void)ws_size;

    const float* depot_xy = (const float*)d_in[0];
    const float* cust_xy  = (const float*)d_in[1];
    const float* demand   = (const float*)d_in[2];
    const float* W_depot  = (const float*)d_in[3];
    const float* b_depot  = (const float*)d_in[4];
    const float* W_init   = (const float*)d_in[5];
    const float* b_init   = (const float*)d_in[6];
    const float* gcn_W    = (const float*)d_in[7];
    const float* gcn_b    = (const float*)d_in[8];
    const float* bn_gamma = (const float*)d_in[9];
    const float* bn_beta  = (const float*)d_in[10];
    const float* ff_W1    = (const float*)d_in[11];
    const float* ff_b1    = (const float*)d_in[12];
    const float* ff_W2    = (const float*)d_in[13];
    const float* ff_b2    = (const float*)d_in[14];

    char* ws = (char*)d_ws;
    size_t off = 0;
    auto walloc = [&](size_t bytes) -> void* {
        void* p = ws + off;
        off = (off + bytes + 255) & ~(size_t)255;
        return p;
    };
    float*  x     = (float*) walloc((size_t)NT_ * C_ * 4);
    float*  hbuf  = (float*) walloc((size_t)NT_ * C_ * 4);
    half_t* xh    = (half_t*)walloc((size_t)NT_ * C_ * 2);
    half_t* f1h   = (half_t*)walloc((size_t)NT_ * H_ * 2);
    float*  stats = (float*) walloc((size_t)2 * C_ * 4);
    half_t* wg    = (half_t*)walloc((size_t)L_ * C_ * C_ * 2);
    half_t* w1    = (half_t*)walloc((size_t)L_ * C_ * H_ * 2);
    half_t* w2    = (half_t*)walloc((size_t)L_ * H_ * C_ * 2);

    // Weight conversion to f16 (B operands, K-major row layout already correct)
    {
        int n;
        n = L_ * C_ * C_;
        cvt_f32_f16_kernel<<<(n + 255) / 256, 256, 0, stream>>>(gcn_W, wg, n);
        n = L_ * C_ * H_;
        cvt_f32_f16_kernel<<<(n + 255) / 256, 256, 0, stream>>>(ff_W1, w1, n);
        n = L_ * H_ * C_;
        cvt_f32_f16_kernel<<<(n + 255) / 256, 256, 0, stream>>>(ff_W2, w2, n);
    }

    // Initial embeddings
    init_embed_kernel<<<(NT_ * C_) / 256, 256, 0, stream>>>(
        depot_xy, cust_xy, demand, W_depot, b_depot, W_init, b_init, x, xh);

    for (int l = 0; l < L_; ++l) {
        // h = x @ gcn_W[l]
        gemm_wmma_kernel<128, 128, 0><<<dim3(200, 2), 256, 0, stream>>>(
            xh, wg + (size_t)l * C_ * C_, nullptr, nullptr, hbuf, nullptr);
        // y = x + agg + b  (closed-form GCN aggregation, in place into hbuf)
        gcn_prefix_kernel<<<B_, C_, 0, stream>>>(hbuf, x, gcn_b + l * C_, hbuf);
        // BN #1
        bn_stats_kernel<<<C_, 256, 0, stream>>>(
            hbuf, bn_gamma + l * C_, bn_beta + l * C_, stats);
        bn_apply_kernel<<<(NT_ * C_) / 256, 256, 0, stream>>>(hbuf, stats, x, xh);
        // FFN
        gemm_wmma_kernel<128, 512, 1><<<dim3(200, 8), 256, 0, stream>>>(
            xh, w1 + (size_t)l * C_ * H_, ff_b1 + l * H_, nullptr, nullptr, f1h);
        gemm_wmma_kernel<512, 128, 2><<<dim3(200, 2), 256, 0, stream>>>(
            f1h, w2 + (size_t)l * H_ * C_, ff_b2 + l * C_, x, hbuf, nullptr);
        // BN #2
        bn_stats_kernel<<<C_, 256, 0, stream>>>(
            hbuf, bn_gamma + l * C_, bn_beta + l * C_, stats);
        bn_apply_kernel<<<(NT_ * C_) / 256, 256, 0, stream>>>(hbuf, stats, x, xh);
    }

    final_out_kernel<<<B_, C_, 0, stream>>>(
        x, (float*)d_out, (float*)d_out + (size_t)NT_ * C_);
}
